// SingleRNN_61632780698292
// MI455X (gfx1250) — matrix operand order repaired and checked by reading, not compiled, not run on hardware
//
#include <hip/hip_runtime.h>
#include <math.h>

// LSTM (B=512, I=512, H=1024, T=64) on gfx1250 using bf16 WMMA (f32 accum).
// Sequential recurrence -> 3 kernels per step, stream-ordered (graph-safe).

typedef __attribute__((ext_vector_type(16))) __bf16          v16bf;
typedef __attribute__((ext_vector_type(8)))  float           v8f;
typedef __attribute__((ext_vector_type(16))) unsigned short  v16u;
typedef __attribute__((ext_vector_type(8)))  unsigned short  v8u;
typedef __attribute__((ext_vector_type(4)))  float           v4f;

#define BSZ   512
#define ISZ   512
#define HSZ   1024
#define TSTP  64
#define KXH   1536   /* I + H : fused GEMM K */
#define NG    4096   /* 4*H   : gate columns */

__device__ __forceinline__ unsigned short f2bf_bits(float f) {
  unsigned x = __float_as_uint(f);
  unsigned r = x + 0x7FFFu + ((x >> 16) & 1u);   // round-to-nearest-even
  return (unsigned short)(r >> 16);
}

// ---- WMMA fragment loaders (wave32, 16x16x32 bf16) -------------------------
// A 16x32 (MxK), 16-bit: lanes 0-15 -> M=lane, K base 0; lanes 16-31 -> M=lane-16, K base 16.
// vector elems 0..15 = K kb..kb+15.
__device__ __forceinline__ v16bf load_frag_a(const unsigned short* A, int lda,
                                             int row0, int k0, int lane) {
  int m  = lane & 15;
  int kb = (lane >> 4) << 4;
  const unsigned short* p = A + (size_t)(row0 + m) * lda + (k0 + kb);
  v8u lo = *(const v8u*)p;
  v8u hi = *(const v8u*)(p + 8);
  v16u t;
#pragma unroll
  for (int i = 0; i < 8; ++i) { t[i] = lo[i]; t[i + 8] = hi[i]; }
  return __builtin_bit_cast(v16bf, t);
}

// B 32x16 (KxN), 16-bit: lanes 0-15 -> N=lane, K=0..15; lanes 16-31 -> N=lane-16, K=16..31.
// Weight stored row-major [N][K] so each lane reads 32 contiguous bytes.
__device__ __forceinline__ v16bf load_frag_b_bf16(const unsigned short* W, int ldb,
                                                  int col0, int k0, int lane) {
  int n  = lane & 15;
  int kb = (lane >> 4) << 4;
  const unsigned short* p = W + (size_t)(col0 + n) * ldb + (k0 + kb);
  v16u t = *(const v16u*)p;
  return __builtin_bit_cast(v16bf, t);
}

// Same B layout but source is f32 (per-step Wl_t, read once -> convert in-register).
__device__ __forceinline__ v16bf load_frag_b_f32(const float* W, int ldb,
                                                 int col0, int k0, int lane) {
  int n  = lane & 15;
  int kb = (lane >> 4) << 4;
  const float* p = W + (size_t)(col0 + n) * ldb + (k0 + kb);
  v4f q0 = ((const v4f*)p)[0];
  v4f q1 = ((const v4f*)p)[1];
  v4f q2 = ((const v4f*)p)[2];
  v4f q3 = ((const v4f*)p)[3];
  v16u t;
#pragma unroll
  for (int i = 0; i < 4; ++i) {
    t[i]      = f2bf_bits(q0[i]);
    t[i + 4]  = f2bf_bits(q1[i]);
    t[i + 8]  = f2bf_bits(q2[i]);
    t[i + 12] = f2bf_bits(q3[i]);
  }
  return __builtin_bit_cast(v16bf, t);
}

// ---- weight pack: [W_ih | W_hh] -> bf16 [4096][1536] -----------------------
__global__ void lstm_pack_weights(const float* __restrict__ W_ih,
                                  const float* __restrict__ W_hh,
                                  unsigned short* __restrict__ Wp) {
  int idx = blockIdx.x * blockDim.x + threadIdx.x;
  if (idx >= NG * KXH) return;
  int n = idx / KXH;
  int k = idx - n * KXH;
  float v = (k < ISZ) ? W_ih[(size_t)n * ISZ + k]
                      : W_hh[(size_t)n * HSZ + (k - ISZ)];
  Wp[idx] = f2bf_bits(v);
}

// ---- state init: xh = [0 | bf16(hidden)], cx = 0 ---------------------------
__global__ void lstm_init_state(const float* __restrict__ hidden,
                                unsigned short* __restrict__ xh,
                                float* __restrict__ cx) {
  int idx = blockIdx.x * blockDim.x + threadIdx.x;
  if (idx < BSZ * KXH) {
    int b = idx / KXH;
    int c = idx - b * KXH;
    xh[idx] = (c < ISZ) ? (unsigned short)0
                        : f2bf_bits(hidden[(size_t)b * HSZ + (c - ISZ)]);
  }
  if (idx < BSZ * HSZ) cx[idx] = 0.0f;
}

// ---- gates GEMM: gates[B][4096] = xh[B][1536] @ Wp^T -----------------------
// Block = 8 waves -> 64(M) x 256(N) tile; wave = 32(M) x 64(N) = 2x4 WMMA accums.
__global__ __launch_bounds__(256)
void lstm_gates_gemm(const unsigned short* __restrict__ xh,
                     const unsigned short* __restrict__ Wp,
                     float* __restrict__ gates) {
  const int lane  = threadIdx.x & 31;
  const int wave  = threadIdx.x >> 5;
  const int mtile = blockIdx.x * 64  + (wave & 1) * 32;
  const int ntile = blockIdx.y * 256 + (wave >> 1) * 64;

  v8f acc[2][4] = {};
  for (int k0 = 0; k0 < KXH; k0 += 32) {
    v16bf a[2], b[4];
#pragma unroll
    for (int i = 0; i < 2; ++i) a[i] = load_frag_a(xh, KXH, mtile + i * 16, k0, lane);
#pragma unroll
    for (int j = 0; j < 4; ++j) b[j] = load_frag_b_bf16(Wp, KXH, ntile + j * 16, k0, lane);
#pragma unroll
    for (int i = 0; i < 2; ++i)
#pragma unroll
      for (int j = 0; j < 4; ++j)
        acc[i][j] = __builtin_amdgcn_wmma_f32_16x16x32_bf16(
            false, a[i], false, b[j], (short)0, acc[i][j], false, false);
  }

  // C/D layout: VGPR r -> M = r + 8*(lane/16); N = lane%16.
  const int mh = (lane >> 4) << 3;
  const int nc = lane & 15;
#pragma unroll
  for (int i = 0; i < 2; ++i)
#pragma unroll
    for (int j = 0; j < 4; ++j) {
      int mb = mtile + i * 16 + mh;
      int nb = ntile + j * 16 + nc;
#pragma unroll
      for (int r = 0; r < 8; ++r)
        gates[(size_t)(mb + r) * NG + nb] = acc[i][j][r];
    }
}

// ---- LSTM cell pointwise: i,f,g,o -> cx, hx (hx written bf16 into xh) ------
__global__ __launch_bounds__(256)
void lstm_pointwise(const float* __restrict__ gates,
                    const float* __restrict__ b_ih,
                    const float* __restrict__ b_hh,
                    float* __restrict__ cx,
                    unsigned short* __restrict__ xh) {
  int idx = blockIdx.x * blockDim.x + threadIdx.x;
  if (idx >= BSZ * HSZ) return;
  int b = idx / HSZ;
  int h = idx - b * HSZ;
  const float* g = gates + (size_t)b * NG;
  float gi = g[h]           + b_ih[h]           + b_hh[h];
  float gf = g[HSZ + h]     + b_ih[HSZ + h]     + b_hh[HSZ + h];
  float gg = g[2 * HSZ + h] + b_ih[2 * HSZ + h] + b_hh[2 * HSZ + h];
  float go = g[3 * HSZ + h] + b_ih[3 * HSZ + h] + b_hh[3 * HSZ + h];
  float i_ = 1.0f / (1.0f + __expf(-gi));
  float f_ = 1.0f / (1.0f + __expf(-gf));
  float g_ = tanhf(gg);
  float o_ = 1.0f / (1.0f + __expf(-go));
  float c  = f_ * cx[idx] + i_ * g_;
  cx[idx]  = c;
  float hx = o_ * tanhf(c);
  xh[(size_t)b * KXH + ISZ + h] = f2bf_bits(hx);
}

// ---- per-step linear: x = hx @ Wl_t^T + bl_t; write f32 out + bf16 x -> xh -
__global__ __launch_bounds__(256)
void lstm_out_gemm(unsigned short* __restrict__ xh,       // read h-part, write x-part
                   const float* __restrict__ Wl_t,        // [I][H] f32
                   const float* __restrict__ bl_t,        // [I]
                   float* __restrict__ out_t) {           // d_out + t*I, row stride T*I
  const int lane  = threadIdx.x & 31;
  const int wave  = threadIdx.x >> 5;
  const int mtile = blockIdx.x * 64  + (wave & 1) * 32;
  const int ntile = blockIdx.y * 256 + (wave >> 1) * 64;

  const unsigned short* A = xh + ISZ;   // hx lives in columns [ISZ, ISZ+HSZ)
  v8f acc[2][4] = {};
  for (int k0 = 0; k0 < HSZ; k0 += 32) {
    v16bf a[2], b[4];
#pragma unroll
    for (int i = 0; i < 2; ++i) a[i] = load_frag_a(A, KXH, mtile + i * 16, k0, lane);
#pragma unroll
    for (int j = 0; j < 4; ++j) b[j] = load_frag_b_f32(Wl_t, HSZ, ntile + j * 16, k0, lane);
#pragma unroll
    for (int i = 0; i < 2; ++i)
#pragma unroll
      for (int j = 0; j < 4; ++j)
        acc[i][j] = __builtin_amdgcn_wmma_f32_16x16x32_bf16(
            false, a[i], false, b[j], (short)0, acc[i][j], false, false);
  }

  const int mh = (lane >> 4) << 3;
  const int nc = lane & 15;
#pragma unroll
  for (int i = 0; i < 2; ++i)
#pragma unroll
    for (int j = 0; j < 4; ++j) {
      int mb = mtile + i * 16 + mh;
      int nb = ntile + j * 16 + nc;
      float bias = bl_t[nb];
#pragma unroll
      for (int r = 0; r < 8; ++r) {
        float v = acc[i][j][r] + bias;
        out_t[(size_t)(mb + r) * ((size_t)TSTP * ISZ) + nb] = v;   // [B,T,I]
        xh[(size_t)(mb + r) * KXH + nb] = f2bf_bits(v);            // x for next step
      }
    }
}

extern "C" void kernel_launch(void* const* d_in, const int* in_sizes, int n_in,
                              void* d_out, int out_size, void* d_ws, size_t ws_size,
                              hipStream_t stream) {
  (void)in_sizes; (void)n_in; (void)out_size; (void)ws_size;
  const float* hidden = (const float*)d_in[0];
  const float* W_ih   = (const float*)d_in[1];
  const float* W_hh   = (const float*)d_in[2];
  const float* b_ih   = (const float*)d_in[3];
  const float* b_hh   = (const float*)d_in[4];
  const float* Wl     = (const float*)d_in[5];
  const float* bl     = (const float*)d_in[6];
  float* out = (float*)d_out;

  // Scratch carve-up (all offsets 4KB-ish aligned):
  char* ws = (char*)d_ws;
  unsigned short* Wp    = (unsigned short*)ws;                          // 4096*1536*2  = 12.6 MB
  unsigned short* xh    = (unsigned short*)(ws + (size_t)NG * KXH * 2); //  512*1536*2  =  1.6 MB
  float*          gates = (float*)(ws + (size_t)NG * KXH * 2
                                      + (size_t)BSZ * KXH * 2);          //  512*4096*4  =  8.4 MB
  float*          cx    = (float*)((char*)gates + (size_t)BSZ * NG * 4); //  512*1024*4  =  2.1 MB

  // 1) pack recurrent weights to bf16 (once per call; re-done every call -> deterministic)
  {
    int total = NG * KXH;
    lstm_pack_weights<<<dim3((total + 255) / 256), dim3(256), 0, stream>>>(W_ih, W_hh, Wp);
  }
  // 2) init xh / cx
  {
    int total = BSZ * KXH;
    lstm_init_state<<<dim3((total + 255) / 256), dim3(256), 0, stream>>>(hidden, xh, cx);
  }
  // 3) 64 recurrent steps, stream-ordered
  for (int t = 0; t < TSTP; ++t) {
    lstm_gates_gemm<<<dim3(BSZ / 64, NG / 256), dim3(256), 0, stream>>>(xh, Wp, gates);
    lstm_pointwise<<<dim3((BSZ * HSZ) / 256), dim3(256), 0, stream>>>(gates, b_ih, b_hh, cx, xh);
    lstm_out_gemm<<<dim3(BSZ / 64, ISZ / 256), dim3(256), 0, stream>>>(
        xh, Wl + (size_t)t * ISZ * HSZ, bl + (size_t)t * ISZ, out + (size_t)t * ISZ);
  }
}
